// up_block11_57458072486024
// MI455X (gfx1250) — matrix operand order repaired and checked by reading, not compiled, not run on hardware
//
#include <hip/hip_runtime.h>

typedef _Float16 f16;
typedef __attribute__((ext_vector_type(16))) _Float16 v16h;
typedef __attribute__((ext_vector_type(8)))  _Float16 v8h;
typedef __attribute__((ext_vector_type(8)))  float    v8f;

#define DEVFN static __device__ __forceinline__

// ---- problem constants ----
#define BB   4
#define CIN  128
#define NP   1024
#define KNN  9
#define COLS 9216      /* N*KNN */
#define M4   4096      /* N*UP  */
#define CP   160       /* attention channels 130 padded to mult of 32 */
#define CHP  144       /* h-GEMM output rows, 130 padded to mult of 16 */

// ===================== WMMA fragment helpers (contiguous k order, same in A and B) ==========
// A 16x32 (MxK), row-major, unit K-stride. lane l: m=(l&15), K = (l>>4)*16 + e
// -> per lane 16 contiguous f16 => two b128 loads.
DEVFN v16h frag_a_rm(const f16* p, long m0, long k0, long lda, int lane) {
  long base = (m0 + (lane & 15)) * lda + k0 + (long)((lane >> 4) * 16);
  v8h lo = *(const v8h*)(p + base);
  v8h hi = *(const v8h*)(p + base + 8);
  return __builtin_shufflevector(lo, hi, 0, 1, 2, 3, 4, 5, 6, 7,
                                 8, 9, 10, 11, 12, 13, 14, 15);
}
// B 32x16 (KxN) with unit K-stride (i.e. source stored N-major): lane l: n=(l&15),
// K = (l>>4)*16 + e -> 16 contiguous f16 per lane => two b128 loads.
DEVFN v16h frag_b_kunit(const f16* p, long k0, long n0, long sn, int lane) {
  long base = (n0 + (lane & 15)) * sn + k0 + (long)((lane >> 4) * 16);
  v8h lo = *(const v8h*)(p + base);
  v8h hi = *(const v8h*)(p + base + 8);
  return __builtin_shufflevector(lo, hi, 0, 1, 2, 3, 4, 5, 6, 7,
                                 8, 9, 10, 11, 12, 13, 14, 15);
}
// B 32x16 fragment from an LDS tile stored K-major ([k*16+n] f16)
DEVFN v16h frag_b_lds(const f16* tile, int lane) {
  int n = lane & 15;
  int kb = (lane >> 4) * 16;
  v16h r;
#pragma unroll
  for (int e = 0; e < 16; ++e) r[e] = tile[(kb + e) * 16 + n];
  return r;
}
DEVFN v8f wmma16(v16h a, v16h b, v8f c) {
  return __builtin_amdgcn_wmma_f32_16x16x32_f16(false, a, false, b, (short)0, c, false, false);
}

// ======= generic row-major f16 WMMA GEMM, async-to-LDS staged B operand ======
// one wave per 16x16 output tile; K multiple of 32.
// B tile (32 rows x 16 cols = 32B/row) staged via global_load_async_to_lds_b128:
// each lane DMAs one 32-byte row straight into LDS (ASYNCcnt), no VGPR roundtrip.
template <bool BIAS, bool RELU>
__global__ void gemm_rm_kernel(const f16* __restrict__ A, long lda, long sAb,
                               const f16* __restrict__ B, long ldb, long sBb,
                               float* __restrict__ C, long ldc, long sCb,
                               int K, const float* __restrict__ bias) {
  __shared__ f16 btile[32 * 16];
  int b  = blockIdx.z;
  int m0 = blockIdx.y * 16;
  int n0 = blockIdx.x * 16;
  int lane = threadIdx.x;
  const f16* Ab = A + (long)b * sAb;
  const f16* Bb = B + (long)b * sBb;
  // low 32 bits of a flat LDS address are the wave-relative LDS byte offset
  unsigned ldsoff = (unsigned)(unsigned long long)(&btile[lane * 16]);
  v8f acc = {};
  for (int k0 = 0; k0 < K; k0 += 32) {
    const f16* gsrc = Bb + (long)(k0 + lane) * ldb + n0;
    asm volatile(
        "s_wait_dscnt 0\n\t"                                // prev frag reads done
        "global_load_async_to_lds_b128 %0, %1, off\n\t"     // row bytes [0,16)
        "global_load_async_to_lds_b128 %0, %1, off offset:16\n\t"  // bytes [16,32)
        "s_wait_asynccnt 0x0"
        :
        : "v"(ldsoff), "v"((unsigned long long)gsrc)
        : "memory");
    v16h af = frag_a_rm(Ab, m0, k0, lda, lane);
    v16h bf = frag_b_lds(btile, lane);
    acc = wmma16(af, bf, acc);
  }
  int col   = n0 + (lane & 15);
  int rbase = m0 + ((lane >> 4) * 8);
#pragma unroll
  for (int r = 0; r < 8; ++r) {
    float v = acc[r];
    if (BIAS) v += bias[rbase + r];
    if (RELU) v = fmaxf(v, 0.f);
    C[(long)b * sCb + (long)(rbase + r) * ldc + col] = v;
  }
}

// ===================== misc elementwise / prep kernels =======================
__global__ void wconv_kernel(const float* __restrict__ W, int M, int Kd, int Mp, int Kp,
                             f16* __restrict__ dst) {
  int i = blockIdx.x * blockDim.x + threadIdx.x;
  if (i >= Mp * Kp) return;
  int k = i % Kp, m = i / Kp;
  float v = (m < M && k < Kd) ? W[(long)m * Kd + k] : 0.f;
  dst[i] = (f16)v;
}

// L2-normalize per point; emit row-major (n,c) for A-operand and (c,n) for B-operand
__global__ void normalize_kernel(const float* __restrict__ x, f16* __restrict__ xn16,
                                 f16* __restrict__ xnT16) {
  int i = blockIdx.x * blockDim.x + threadIdx.x;   // b*NP + n
  if (i >= BB * NP) return;
  int b = i / NP, n = i % NP;
  const float* p = x + (long)b * CIN * NP + n;
  float s = 0.f;
  for (int c = 0; c < CIN; ++c) { float v = p[(long)c * NP]; s += v * v; }
  float inv = 1.f / fmaxf(sqrtf(s), 1e-12f);
  f16* q  = xn16 + (long)i * CIN;
  f16* qt = xnT16 + (long)b * CIN * NP + n;
  for (int c = 0; c < CIN; ++c) {
    f16 v = (f16)(p[(long)c * NP] * inv);
    q[c] = v;
    qt[(long)c * NP] = v;
  }
}

// per (b,n): extract 18 largest gram entries (nearest neighbors), ordered, tie->lowest idx
__global__ void topk18_kernel(const float* __restrict__ gram, int* __restrict__ idx18) {
  int bn = blockIdx.x;                  // b*NP + n
  int lane = threadIdx.x;               // 32 lanes
  const float* row = gram + (long)bn * NP;
  unsigned sel = 0;                     // bitmask over this lane's 32 candidates
  for (int t = 0; t < 18; ++t) {
    float bestv = -3.0e38f; int besti = 1 << 30;
    for (int j = 0; j < 32; ++j) {
      if (sel & (1u << j)) continue;
      int m = j * 32 + lane;
      float v = row[m];
      if (v > bestv || (v == bestv && m < besti)) { bestv = v; besti = m; }
    }
    for (int off = 16; off > 0; off >>= 1) {
      float ov = __shfl_xor(bestv, off, 32);
      int   oi = __shfl_xor(besti, off, 32);
      if (ov > bestv || (ov == bestv && oi < besti)) { bestv = ov; besti = oi; }
    }
    if ((besti & 31) == lane) sel |= 1u << (besti >> 5);
    if (lane == 0) idx18[(long)bn * 18 + t] = besti;
  }
}

// build [x_i ; x_j - x_i] feature matrix (B, 256, COLS) in f16
__global__ void feat_kernel(const float* __restrict__ x, const int* __restrict__ idx18,
                            int stridej, f16* __restrict__ feat) {
  int i = blockIdx.x * blockDim.x + threadIdx.x;   // b*COLS + col
  if (i >= BB * COLS) return;
  int b = i / COLS, col = i % COLS;
  int n = col / KNN, j = col % KNN;
  int nb = idx18[((long)b * NP + n) * 18 + j * stridej];
  const float* xb = x + (long)b * CIN * NP;
  f16* fo = feat + (long)b * (2 * CIN) * COLS + col;
  for (int c = 0; c < CIN; ++c) {
    float xi = xb[(long)c * NP + n];
    float xj = xb[(long)c * NP + nb];
    fo[(long)c * COLS] = (f16)xi;
    fo[(long)(c + CIN) * COLS] = (f16)(xj - xi);
  }
}

// per-channel sum / sumsq over all batches (training-mode BN stats)
__global__ void stats_kernel(const float* __restrict__ X, long bstride, long rstride,
                             long cols, float* __restrict__ sums, float* __restrict__ sqs) {
  int c = blockIdx.x;
  int tid = threadIdx.x;
  float s = 0.f, q = 0.f;
  for (int b = 0; b < BB; ++b) {
    const float* p = X + (long)b * bstride + (long)c * rstride;
    for (long i = tid; i < cols; i += blockDim.x) { float v = p[i]; s += v; q += v * v; }
  }
  __shared__ float sh[256], shq[256];
  sh[tid] = s; shq[tid] = q; __syncthreads();
  for (int o = blockDim.x >> 1; o > 0; o >>= 1) {
    if (tid < o) { sh[tid] += sh[tid + o]; shq[tid] += shq[tid + o]; }
    __syncthreads();
  }
  if (tid == 0) { sums[c] = sh[0]; sqs[c] = shq[0]; }
}

// edge conv tail: bn -> relu -> max over k=9
__global__ void edge_final_kernel(const float* __restrict__ out, const float* __restrict__ sums,
                                  const float* __restrict__ sqs, const float* __restrict__ ga,
                                  const float* __restrict__ be, float* __restrict__ xout) {
  long i = (long)blockIdx.x * blockDim.x + threadIdx.x;  // b*256*NP
  if (i >= (long)BB * 256 * NP) return;
  int n = (int)(i % NP); int c = (int)((i / NP) % 256); int b = (int)(i / ((long)NP * 256));
  float cnt = (float)BB * COLS;
  float mean = sums[c] / cnt;
  float var  = sqs[c] / cnt - mean * mean;
  float rs = rsqrtf(var + 1e-5f) * ga[c];
  float sh = be[c];
  const float* p = out + ((long)b * 256 + c) * COLS + (long)n * KNN;
  float mx = 0.f;   // relu outputs are >= 0
  for (int j = 0; j < KNN; ++j) {
    float v = fmaxf((p[j] - mean) * rs + sh, 0.f);
    mx = fmaxf(mx, v);
  }
  xout[i] = mx;
}

// build (B, CP=160, 4096) f16 net matrix: rows 0..127 from reshaped x1/x2, 128..129 grid, rest 0
__global__ void net_kernel(const float* __restrict__ x1, const float* __restrict__ x2,
                           f16* __restrict__ net16) {
  long i = (long)blockIdx.x * blockDim.x + threadIdx.x;
  if (i >= (long)BB * CP * M4) return;
  int m = (int)(i % M4); int c = (int)((i / M4) % CP); int b = (int)(i / ((long)M4 * CP));
  float v = 0.f;
  if (c < 128) {
    int u = m & 3, n = m >> 2;
    int cc = u * 128 + c;               // channel in concat(x1,x2) of 512
    v = (cc < 256) ? x1[((long)b * 256 + cc) * NP + n]
                   : x2[((long)b * 256 + (cc - 256)) * NP + n];
  } else if (c < 130) {
    int gi = m >> 10;                   // repeat(grid, N): row = m // 1024
    float gx = (gi < 2) ? -0.2f : 0.2f;
    float gy = (gi & 1) ? 0.2f : -0.2f;
    v = (c == 128) ? gx : gy;
  }
  net16[i] = (f16)v;
}

// bn (training stats) -> relu -> f16; transposed=1 writes (cols x Cpad) layout
__global__ void bn_relu_f16_kernel(const float* __restrict__ X, long bstride, long rstride,
                                   const float* __restrict__ sums, const float* __restrict__ sqs,
                                   float cnt, const float* __restrict__ ga,
                                   const float* __restrict__ be, int Cvalid, int Cpad,
                                   long cols, int transposed, f16* __restrict__ dst) {
  long i = (long)blockIdx.x * blockDim.x + threadIdx.x;
  if (i >= (long)BB * Cpad * cols) return;
  long m = i % cols; int c = (int)((i / cols) % Cpad); int b = (int)(i / (cols * Cpad));
  float v = 0.f;
  if (c < Cvalid) {
    float mean = sums[c] / cnt;
    float var  = sqs[c] / cnt - mean * mean;
    float x = X[(long)b * bstride + (long)c * rstride + m];
    v = fmaxf((x - mean) * rsqrtf(var + 1e-5f) * ga[c] + be[c], 0.f);
  }
  long o = transposed ? (((long)b * cols + m) * Cpad + c)
                      : (((long)b * Cpad + c) * cols + m);
  dst[o] = (f16)v;
}

__global__ void f32_to_f16_kernel(const float* __restrict__ src, f16* __restrict__ dst, long total) {
  long i = (long)blockIdx.x * blockDim.x + threadIdx.x;
  if (i < total) dst[i] = (f16)src[i];
}

// ========= attention pass A: streaming softmax stats (rowmax, denom) per n ==
// one wave per (b, 16-row n-tile); s[n,m] = sum_c g[c,n]*f[c,m], K=32.
// gT/fT stored (M4 x 32) so every fragment is two b128 loads.
__global__ void attn_stats_kernel(const f16* __restrict__ gT, const f16* __restrict__ fT,
                                  float* __restrict__ rowmax, float* __restrict__ denom) {
  int b = blockIdx.y;
  int n0 = blockIdx.x * 16;
  int lane = threadIdx.x;
  const f16* gb = gT + (long)b * M4 * 32;
  const f16* fb = fT + (long)b * M4 * 32;
  v16h af = frag_a_rm(gb, n0, 0, 32, lane);          // A(n,c) = gT[n*32 + c]
  float mx[8], sm[8];
#pragma unroll
  for (int r = 0; r < 8; ++r) { mx[r] = -3.0e38f; sm[r] = 0.f; }
  for (int mt = 0; mt < M4 / 16; ++mt) {
    v16h bf = frag_b_kunit(fb, 0, mt * 16, 32, lane); // B(c,m) = fT[m*32 + c]
    v8f s = {};
    s = wmma16(af, bf, s);
#pragma unroll
    for (int r = 0; r < 8; ++r) {
      float v = s[r];
      float nm = fmaxf(mx[r], v);
      sm[r] = sm[r] * __expf(mx[r] - nm) + __expf(v - nm);
      mx[r] = nm;
    }
  }
  // combine across the 16 lanes holding the same rows
  for (int off = 1; off < 16; off <<= 1) {
#pragma unroll
    for (int r = 0; r < 8; ++r) {
      float omx = __shfl_xor(mx[r], off, 32);
      float osm = __shfl_xor(sm[r], off, 32);
      float nm = fmaxf(mx[r], omx);
      sm[r] = sm[r] * __expf(mx[r] - nm) + osm * __expf(omx - nm);
      mx[r] = nm;
    }
  }
  if ((lane & 15) == 0) {
    int rbase = n0 + ((lane >> 4) * 8);
#pragma unroll
    for (int r = 0; r < 8; ++r) {
      rowmax[(long)b * M4 + rbase + r] = mx[r];
      denom[(long)b * M4 + rbase + r]  = sm[r];
    }
  }
}

// ========= attention pass B: o = h @ softmax(s), epilogue gamma*o + net ======
// one wave per (b, 16-col m-tile); recompute s tiles, weights via LDS -> f16 B-frag
__global__ void attn_apply_kernel(const f16* __restrict__ gT, const f16* __restrict__ fT,
                                  const f16* __restrict__ h16, const f16* __restrict__ net16,
                                  const float* __restrict__ rowmax, const float* __restrict__ denom,
                                  const float* __restrict__ gamma, f16* __restrict__ attn16) {
  __shared__ f16 wlds[32 * 16];
  int b = blockIdx.y;
  int m0 = blockIdx.x * 16;
  int lane = threadIdx.x;
  const f16* gb = gT + (long)b * M4 * 32;
  const f16* fb = fT + (long)b * M4 * 32;
  const f16* hb = h16 + (long)b * CP * M4;
  const float* rmax = rowmax + (long)b * M4;
  const float* den  = denom  + (long)b * M4;
  v16h bf_f = frag_b_kunit(fb, 0, m0, 32, lane);   // constant across k-loop
  v8f acc[10];
#pragma unroll
  for (int t = 0; t < 10; ++t) acc[t] = (v8f){};
  int rowoff = (lane < 16) ? 0 : 8;
  int col = lane & 15;
  for (int kt = 0; kt < M4 / 32; ++kt) {
    int n0 = kt * 32;
#pragma unroll
    for (int half = 0; half < 2; ++half) {
      v16h ag = frag_a_rm(gb, n0 + 16 * half, 0, 32, lane);
      v8f s = {};
      s = wmma16(ag, bf_f, s);
#pragma unroll
      for (int r = 0; r < 8; ++r) {
        int n = n0 + 16 * half + rowoff + r;
        float w = __expf(s[r] - rmax[n]) / den[n];
        wlds[(16 * half + rowoff + r) * 16 + col] = (f16)w;
      }
    }
    __syncthreads();
    v16h wf = frag_b_lds(wlds, lane);
#pragma unroll
    for (int ct = 0; ct < 10; ++ct) {
      v16h ah = frag_a_rm(hb, ct * 16, n0, M4, lane);  // A(c,n) = h[c*M4 + n]
      acc[ct] = wmma16(ah, wf, acc[ct]);
    }
    __syncthreads();
  }
  float gm = gamma[0];
#pragma unroll
  for (int ct = 0; ct < 10; ++ct) {
#pragma unroll
    for (int r = 0; r < 8; ++r) {
      int c = ct * 16 + rowoff + r;
      long off = (long)b * CP * M4 + (long)c * M4 + m0 + col;
      float val = gm * acc[ct][r] + (float)net16[off];
      attn16[off] = (f16)val;
    }
  }
}

// ============================= host orchestration ============================
extern "C" void kernel_launch(void* const* d_in, const int* in_sizes, int n_in,
                              void* d_out, int out_size, void* d_ws, size_t ws_size,
                              hipStream_t stream) {
  (void)in_sizes; (void)n_in; (void)out_size; (void)ws_size;
  const float* inp  = (const float*)d_in[0];
  const float* Wa   = (const float*)d_in[1];
  const float* gna  = (const float*)d_in[2];
  const float* bta  = (const float*)d_in[3];
  const float* Wb   = (const float*)d_in[4];
  const float* gnb  = (const float*)d_in[5];
  const float* btb  = (const float*)d_in[6];
  const float* Wf   = (const float*)d_in[7];
  const float* gf   = (const float*)d_in[9];
  const float* bef  = (const float*)d_in[10];
  const float* Wg   = (const float*)d_in[11];
  const float* gg   = (const float*)d_in[13];
  const float* beg  = (const float*)d_in[14];
  const float* Wh   = (const float*)d_in[15];
  const float* gh   = (const float*)d_in[17];
  const float* beh  = (const float*)d_in[18];
  const float* gam  = (const float*)d_in[19];
  const float* W1   = (const float*)d_in[20];
  const float* b1   = (const float*)d_in[21];
  const float* W2   = (const float*)d_in[22];
  const float* b2   = (const float*)d_in[23];
  float* out = (float*)d_out;

  char* ws = (char*)d_ws;
  size_t cur = 0;
  auto alloc = [&](size_t bytes) -> char* {
    char* p = ws + cur;
    cur += (bytes + 255) & ~(size_t)255;
    return p;
  };
  // weights (f16, zero-padded)
  f16* Wa16 = (f16*)alloc(256 * 256 * 2);
  f16* Wb16 = (f16*)alloc(256 * 256 * 2);
  f16* Wf16 = (f16*)alloc(32 * CP * 2);
  f16* Wg16 = (f16*)alloc(32 * CP * 2);
  f16* Wh16 = (f16*)alloc((size_t)CHP * CP * 2);
  f16* W116 = (f16*)alloc(256 * CP * 2);
  f16* W216 = (f16*)alloc(128 * 256 * 2);
  // data buffers
  f16*   xn16  = (f16*)alloc((size_t)BB * NP * CIN * 2);   // (n,c) A-operand
  f16*   xnT16 = (f16*)alloc((size_t)BB * CIN * NP * 2);   // (c,n) B-operand
  int*   idx18 = (int*)alloc((size_t)BB * NP * 18 * 4);
  f16*   feat  = (f16*)alloc((size_t)BB * 256 * COLS * 2);
  // big reused f32 scratch region (gram -> edge gemm out -> f/g/h/out1)
  char*  S1    = alloc((size_t)BB * 256 * COLS * 4);           // 37.75 MB
  float* gram  = (float*)S1;                                   // B*1024*1024
  float* eout  = (float*)S1;                                   // B*256*9216
  float* fOut  = (float*)S1;                                   // B*32*4096
  float* gOut  = (float*)(S1 + (size_t)BB * 32 * M4 * 4);
  float* hOut  = (float*)(S1 + (size_t)2 * BB * 32 * M4 * 4);
  float* out1  = (float*)(S1 + (size_t)2 * BB * 32 * M4 * 4 + (size_t)BB * CHP * M4 * 4);
  float* x1    = (float*)alloc((size_t)BB * 256 * NP * 4);
  float* x2    = (float*)alloc((size_t)BB * 256 * NP * 4);
  f16*   net16 = (f16*)alloc((size_t)BB * CP * M4 * 2);
  f16*   fT16  = (f16*)alloc((size_t)BB * M4 * 32 * 2);    // (m,c) transposed
  f16*   gT16  = (f16*)alloc((size_t)BB * M4 * 32 * 2);    // (n,c) transposed
  f16*   h16   = (f16*)alloc((size_t)BB * CP * M4 * 2);
  float* rowmx = (float*)alloc((size_t)BB * M4 * 4);
  float* denom = (float*)alloc((size_t)BB * M4 * 4);
  f16*   attn16 = (f16*)alloc((size_t)BB * CP * M4 * 2);
  f16*   out116 = (f16*)alloc((size_t)BB * 256 * M4 * 2);
  float* sums  = (float*)alloc(256 * 4);
  float* sqs   = (float*)alloc(256 * 4);

  auto wconv = [&](const float* src, int M, int Kd, int Mp, int Kp, f16* dst) {
    int total = Mp * Kp;
    wconv_kernel<<<(total + 255) / 256, 256, 0, stream>>>(src, M, Kd, Mp, Kp, dst);
  };
  auto gemm = [&](const f16* A, long lda, long sAb,
                  const f16* B, long ldb, long sBb,
                  float* C, long ldc, long sCb, int M, int Nc, int K,
                  const float* bias, bool relu) {
    dim3 grid(Nc / 16, M / 16, BB);
    if (bias && relu)
      gemm_rm_kernel<true, true><<<grid, 32, 0, stream>>>(A, lda, sAb, B, ldb, sBb,
                                                          C, ldc, sCb, K, bias);
    else
      gemm_rm_kernel<false, false><<<grid, 32, 0, stream>>>(A, lda, sAb, B, ldb, sBb,
                                                            C, ldc, sCb, K, nullptr);
  };

  // --- weights to f16 (zero-padded) ---
  wconv(Wa, 256, 256, 256, 256, Wa16);
  wconv(Wb, 256, 256, 256, 256, Wb16);
  wconv(Wf, 32, 130, 32, CP, Wf16);
  wconv(Wg, 32, 130, 32, CP, Wg16);
  wconv(Wh, 130, 130, CHP, CP, Wh16);
  wconv(W1, 256, 130, 256, CP, W116);
  wconv(W2, 128, 256, 128, 256, W216);

  // --- KNN: normalize, Gram (WMMA), top-18 ---
  normalize_kernel<<<(BB * NP + 255) / 256, 256, 0, stream>>>(inp, xn16, xnT16);
  gemm(xn16, CIN, (long)NP * CIN,
       xnT16, NP, (long)CIN * NP,
       gram, NP, (long)NP * NP, NP, NP, CIN, nullptr, false);
  topk18_kernel<<<BB * NP, 32, 0, stream>>>(gram, idx18);

  // --- edge conv A (dilation 1) ---
  feat_kernel<<<(BB * COLS + 255) / 256, 256, 0, stream>>>(inp, idx18, 1, feat);
  gemm(Wa16, 256, 0, feat, COLS, (long)256 * COLS,
       eout, COLS, (long)256 * COLS, 256, COLS, 256, nullptr, false);
  stats_kernel<<<256, 256, 0, stream>>>(eout, (long)256 * COLS, COLS, COLS, sums, sqs);
  edge_final_kernel<<<(int)(((long)BB * 256 * NP + 255) / 256), 256, 0, stream>>>(
      eout, sums, sqs, gna, bta, x1);

  // --- edge conv B (dilation 2) ---
  feat_kernel<<<(BB * COLS + 255) / 256, 256, 0, stream>>>(inp, idx18, 2, feat);
  gemm(Wb16, 256, 0, feat, COLS, (long)256 * COLS,
       eout, COLS, (long)256 * COLS, 256, COLS, 256, nullptr, false);
  stats_kernel<<<256, 256, 0, stream>>>(eout, (long)256 * COLS, COLS, COLS, sums, sqs);
  edge_final_kernel<<<(int)(((long)BB * 256 * NP + 255) / 256), 256, 0, stream>>>(
      eout, sums, sqs, gnb, btb, x2);

  // --- build net (B,160,4096) f16, rows 128/129 = grid, 130.. zero ---
  net_kernel<<<(int)(((long)BB * CP * M4 + 255) / 256), 256, 0, stream>>>(x1, x2, net16);

  // --- f, g, h conv1d GEMMs + BN + relu (biases absorbed by BN mean) ---
  gemm(Wf16, CP, 0, net16, M4, (long)CP * M4,
       fOut, M4, (long)32 * M4, 32, M4, CP, nullptr, false);
  stats_kernel<<<32, 256, 0, stream>>>(fOut, (long)32 * M4, M4, M4, sums, sqs);
  bn_relu_f16_kernel<<<(int)(((long)BB * 32 * M4 + 255) / 256), 256, 0, stream>>>(
      fOut, (long)32 * M4, M4, sums, sqs, (float)(BB * M4), gf, bef, 32, 32, M4, 1, fT16);

  gemm(Wg16, CP, 0, net16, M4, (long)CP * M4,
       gOut, M4, (long)32 * M4, 32, M4, CP, nullptr, false);
  stats_kernel<<<32, 256, 0, stream>>>(gOut, (long)32 * M4, M4, M4, sums, sqs);
  bn_relu_f16_kernel<<<(int)(((long)BB * 32 * M4 + 255) / 256), 256, 0, stream>>>(
      gOut, (long)32 * M4, M4, sums, sqs, (float)(BB * M4), gg, beg, 32, 32, M4, 1, gT16);

  gemm(Wh16, CP, 0, net16, M4, (long)CP * M4,
       hOut, M4, (long)CHP * M4, CHP, M4, CP, nullptr, false);
  stats_kernel<<<130, 256, 0, stream>>>(hOut, (long)CHP * M4, M4, M4, sums, sqs);
  bn_relu_f16_kernel<<<(int)(((long)BB * CP * M4 + 255) / 256), 256, 0, stream>>>(
      hOut, (long)CHP * M4, M4, sums, sqs, (float)(BB * M4), gh, beh, 130, CP, M4, 0, h16);

  // --- attention: streaming softmax over the 4096x4096 s-matrix (never materialized) ---
  {
    dim3 grid(M4 / 16, BB);
    attn_stats_kernel<<<grid, 32, 0, stream>>>(gT16, fT16, rowmx, denom);
    attn_apply_kernel<<<grid, 32, 0, stream>>>(gT16, fT16, h16, net16, rowmx, denom,
                                               gam, attn16);
  }

  // --- final conv1ds: 130->256 (+b1, relu), 256->128 (+b2, relu) -> d_out ---
  gemm(W116, CP, 0, attn16, M4, (long)CP * M4,
       out1, M4, (long)256 * M4, 256, M4, CP, b1, true);
  f32_to_f16_kernel<<<(int)(((long)BB * 256 * M4 + 255) / 256), 256, 0, stream>>>(
      out1, out116, (long)BB * 256 * M4);
  gemm(W216, 256, 0, out116, M4, (long)256 * M4,
       out, M4, (long)128 * M4, 128, M4, 256, b2, true);
}
